// MSARowAttentionV43_4861902979619
// MI455X (gfx1250) — matrix operand order repaired and checked by reading, not compiled, not run on hardware
//
#include <hip/hip_runtime.h>
#include <hip/hip_bf16.h>

// ---------------------------------------------------------------------------
// Types
// ---------------------------------------------------------------------------
typedef _Float16 v16h __attribute__((ext_vector_type(16)));
typedef _Float16 v8h  __attribute__((ext_vector_type(8)));
typedef float    v8f  __attribute__((ext_vector_type(8)));
typedef float    f4v  __attribute__((ext_vector_type(4)));

#define ATTN_SCALE 0.17677669529663687f   // 32^-0.5
#define LN_EPS_C   1e-5f

// ---------------------------------------------------------------------------
// Helpers
// ---------------------------------------------------------------------------
static __device__ __forceinline__ void f2hl(float x, _Float16& hi, _Float16& lo) {
    _Float16 h = (_Float16)x;
    hi = h;
    lo = (_Float16)(x - (float)h);
}

static __device__ __forceinline__ v16h cat8(v8h a, v8h b) {
    return __builtin_shufflevector(a, b, 0,1,2,3,4,5,6,7,8,9,10,11,12,13,14,15);
}

// B-fragment: halves 0..7 at p+0..7, halves 8..15 at p+16..23; p must be 16B aligned
static __device__ __forceinline__ v16h ld_contig16(const _Float16* p) {
    v8h a = *(const v8h*)p;
    v8h b = *(const v8h*)(p + 16);
    return cat8(a, b);
}

// A-fragment: halves 0..7 at p+0..7 (K = 8*Lh + i), halves 8..15 at p+16..23
static __device__ __forceinline__ v16h ld_a_frag(const _Float16* p) {
    v8h a = *(const v8h*)p;
    v8h b = *(const v8h*)(p + 16);
    return cat8(a, b);
}

// Split-precision fp32-accurate MMA: C += Ah*Bh + Ah*Bl + Al*Bh
static __device__ __forceinline__ v8f wmma3(v16h ah, v16h al, v16h bh, v16h bl, v8f c) {
    c = __builtin_amdgcn_wmma_f32_16x16x32_f16(false, ah, false, bh, (short)0, c, false, false);
    c = __builtin_amdgcn_wmma_f32_16x16x32_f16(false, ah, false, bl, (short)0, c, false, false);
    c = __builtin_amdgcn_wmma_f32_16x16x32_f16(false, al, false, bh, (short)0, c, false, false);
    return c;
}

// ---------------------------------------------------------------------------
// Kernel 1: QKV projection.  out[32768][768] = msa[32768][256] @ W[256][768] + b
// Block: 256 thr (8 waves), computes 128 rows x 64 cols.  Grid (256, 12).
// ---------------------------------------------------------------------------
__global__ __launch_bounds__(256) void msa_qkv_gemm(
    const float* __restrict__ A, const float* __restrict__ W,
    const float* __restrict__ bias, float* __restrict__ out)
{
    __shared__ _Float16 Ah[128 * 40];
    __shared__ _Float16 Al[128 * 40];
    __shared__ _Float16 Wh[64 * 40];
    __shared__ _Float16 Wl[64 * 40];

    const int tid = threadIdx.x;
    const int Lm  = tid & 15;
    const int Lh  = (tid >> 4) & 1;
    const int w   = tid >> 5;
    const int m0  = blockIdx.x * 128;
    const int n0  = blockIdx.y * 64;

    v8f acc[4];
#pragma unroll
    for (int ct = 0; ct < 4; ++ct) acc[ct] = (v8f){0,0,0,0,0,0,0,0};

    for (int kc = 0; kc < 256; kc += 32) {
        // Stage A chunk [128][32] as hi/lo halves
        {
            const int arow = tid & 127;
            const int ks   = (tid >> 7) * 16;
            const float* ap = A + (size_t)(m0 + arow) * 256 + kc + ks;
            v8f a0 = *(const v8f*)ap;
            v8f a1 = *(const v8f*)(ap + 8);
#pragma unroll
            for (int j = 0; j < 8; ++j) {
                f2hl(a0[j], Ah[arow * 40 + ks + j],     Al[arow * 40 + ks + j]);
                f2hl(a1[j], Ah[arow * 40 + ks + 8 + j], Al[arow * 40 + ks + 8 + j]);
            }
        }
        // Stage W chunk transposed: Wt[n][k]
        {
            const int nn = tid & 63;
            const int ks = (tid >> 6) * 8;
#pragma unroll
            for (int j = 0; j < 8; ++j) {
                float v = W[(size_t)(kc + ks + j) * 768 + n0 + nn];
                f2hl(v, Wh[nn * 40 + ks + j], Wl[nn * 40 + ks + j]);
            }
        }
        __syncthreads();

        const _Float16* arp = Ah + (w * 16 + Lm) * 40 + 8 * Lh;
        const _Float16* alp = Al + (w * 16 + Lm) * 40 + 8 * Lh;
        v16h fah = ld_a_frag(arp);
        v16h fal = ld_a_frag(alp);
#pragma unroll
        for (int ct = 0; ct < 4; ++ct) {
            const _Float16* bhp = Wh + (ct * 16 + Lm) * 40 + 8 * Lh;
            const _Float16* blp = Wl + (ct * 16 + Lm) * 40 + 8 * Lh;
            acc[ct] = wmma3(fah, fal, ld_contig16(bhp), ld_contig16(blp), acc[ct]);
        }
        __syncthreads();
    }

#pragma unroll
    for (int ct = 0; ct < 4; ++ct) {
        const int n = n0 + ct * 16 + Lm;
        const float b = bias[n];
#pragma unroll
        for (int r = 0; r < 8; ++r) {
            const int m = m0 + w * 16 + r + 8 * Lh;
            out[(size_t)m * 768 + n] = acc[ct][r] + b;
        }
    }
}

// ---------------------------------------------------------------------------
// Kernel 2: attention.  One block per (s,h).  Block 256 thr (8 waves).
// Dynamic LDS layout (bytes):
//   maskb : 0      .. 65536    (256*256 u8)
//   Khi   : 65536  .. 86016    ([256][40] halves)
//   Klo   : 86016  .. 106496
//   Vth   : 106496 .. 123392   ([32][264] halves, V transposed)
//   Vtl   : 123392 .. 140288
//   P     : 140288 .. 275456   (8 waves x ([16][264] hi + [16][264] lo))
// ---------------------------------------------------------------------------
#define ATTN_SMEM 275456

__global__ __launch_bounds__(256) void msa_attn(
    const float* __restrict__ qkv, const int* __restrict__ mask,
    float* __restrict__ outT)
{
    extern __shared__ char smem[];
    unsigned char* maskb = (unsigned char*)smem;
    _Float16* Khi = (_Float16*)(smem + 65536);
    _Float16* Klo = (_Float16*)(smem + 86016);
    _Float16* Vth = (_Float16*)(smem + 106496);
    _Float16* Vtl = (_Float16*)(smem + 123392);
    _Float16* Pregion = (_Float16*)(smem + 140288);

    const int tid = threadIdx.x;
    const int Lm  = tid & 15;
    const int Lh  = (tid >> 4) & 1;
    const int w   = tid >> 5;
    const int s   = blockIdx.x >> 3;
    const int h   = blockIdx.x & 7;

    // Cooperative staging
    for (int i = tid; i < 65536; i += 256)
        maskb[i] = (unsigned char)(mask[i] != 0);
    for (int i = tid; i < 8192; i += 256) {
        const int n = i >> 5, d = i & 31;
        float kv = qkv[((size_t)(s * 256 + n) * 3 + 1) * 256 + h * 32 + d];
        f2hl(kv, Khi[n * 40 + d], Klo[n * 40 + d]);
        float vv = qkv[((size_t)(s * 256 + n) * 3 + 2) * 256 + h * 32 + d];
        f2hl(vv, Vth[d * 264 + n], Vtl[d * 264 + n]);
    }
    __syncthreads();

    _Float16* Ph = Pregion + (size_t)w * (2 * 16 * 264);
    _Float16* Pl = Ph + 16 * 264;

    for (int strip = w; strip < 16; strip += 8) {
        const int nbase = strip * 16;

        // Q A-fragment directly from global (hi/lo split in registers)
        const float* qp = qkv + ((size_t)(s * 256 + nbase + Lm) * 3) * 256 + h * 32 + 8 * Lh;
        v8f q0 = *(const v8f*)qp;
        v8f q1 = *(const v8f*)(qp + 16);
        v16h fah, fal;
#pragma unroll
        for (int j = 0; j < 8; ++j) {
            _Float16 hi, lo;
            f2hl(q0[j], hi, lo); fah[j] = hi;     fal[j] = lo;
            f2hl(q1[j], hi, lo); fah[j + 8] = hi; fal[j + 8] = lo;
        }

        // S strip = q @ k^T : 16 rows x 256 cols, in 16 accumulator tiles
        v8f acc[16];
#pragma unroll
        for (int t = 0; t < 16; ++t) {
            const _Float16* bhp = Khi + (t * 16 + Lm) * 40 + 8 * Lh;
            const _Float16* blp = Klo + (t * 16 + Lm) * 40 + 8 * Lh;
            v8f z = (v8f){0,0,0,0,0,0,0,0};
            acc[t] = wmma3(fah, fal, ld_contig16(bhp), ld_contig16(blp), z);
        }

        // scale + mask + row max
        float rmax[8];
#pragma unroll
        for (int r = 0; r < 8; ++r) rmax[r] = -3.0e38f;
#pragma unroll
        for (int t = 0; t < 16; ++t) {
            const int m = t * 16 + Lm;
#pragma unroll
            for (int r = 0; r < 8; ++r) {
                const int n = nbase + r + 8 * Lh;
                float sv = acc[t][r] * ATTN_SCALE;
                if (maskb[n * 256 + m] == 0) sv = -1.0e9f;
                acc[t][r] = sv;
                rmax[r] = fmaxf(rmax[r], sv);
            }
        }
#pragma unroll
        for (int r = 0; r < 8; ++r) {
            rmax[r] = fmaxf(rmax[r], __shfl_xor(rmax[r], 1));
            rmax[r] = fmaxf(rmax[r], __shfl_xor(rmax[r], 2));
            rmax[r] = fmaxf(rmax[r], __shfl_xor(rmax[r], 4));
            rmax[r] = fmaxf(rmax[r], __shfl_xor(rmax[r], 8));
        }

        // exp + row sum
        float rsum[8] = {0,0,0,0,0,0,0,0};
#pragma unroll
        for (int t = 0; t < 16; ++t) {
#pragma unroll
            for (int r = 0; r < 8; ++r) {
                float e = __expf(acc[t][r] - rmax[r]);
                acc[t][r] = e;
                rsum[r] += e;
            }
        }
#pragma unroll
        for (int r = 0; r < 8; ++r) {
            rsum[r] += __shfl_xor(rsum[r], 1);
            rsum[r] += __shfl_xor(rsum[r], 2);
            rsum[r] += __shfl_xor(rsum[r], 4);
            rsum[r] += __shfl_xor(rsum[r], 8);
            rsum[r] = 1.0f / rsum[r];
        }

        // Normalize and write P (hi/lo) to wave-private LDS
#pragma unroll
        for (int t = 0; t < 16; ++t) {
            const int m = t * 16 + Lm;
#pragma unroll
            for (int r = 0; r < 8; ++r) {
                const int q = r + 8 * Lh;
                float p = acc[t][r] * rsum[r];
                _Float16 hi, lo;
                f2hl(p, hi, lo);
                Ph[q * 264 + m] = hi;
                Pl[q * 264 + m] = lo;
            }
        }

        // O strip = P @ V : 16 rows x 32 cols
        v8f o[2];
        o[0] = (v8f){0,0,0,0,0,0,0,0};
        o[1] = (v8f){0,0,0,0,0,0,0,0};
#pragma unroll
        for (int kc = 0; kc < 8; ++kc) {
            const _Float16* pah = Ph + Lm * 264 + kc * 32 + 8 * Lh;
            const _Float16* pal = Pl + Lm * 264 + kc * 32 + 8 * Lh;
            v16h fpa = ld_a_frag(pah);
            v16h fpl = ld_a_frag(pal);
#pragma unroll
            for (int dt = 0; dt < 2; ++dt) {
                const _Float16* vhp = Vth + (dt * 16 + Lm) * 264 + kc * 32 + 8 * Lh;
                const _Float16* vlp = Vtl + (dt * 16 + Lm) * 264 + kc * 32 + 8 * Lh;
                o[dt] = wmma3(fpa, fpl, ld_contig16(vhp), ld_contig16(vlp), o[dt]);
            }
        }

        // Store O to tmp[s][n][h][d]
#pragma unroll
        for (int dt = 0; dt < 2; ++dt) {
            const int d = dt * 16 + Lm;
#pragma unroll
            for (int r = 0; r < 8; ++r) {
                const int n = nbase + r + 8 * Lh;
                outT[(size_t)(s * 256 + n) * 256 + h * 32 + d] = o[dt][r];
            }
        }
    }
}

// ---------------------------------------------------------------------------
// Kernel 3: out @ W_out + b_out + residual + LayerNorm, fused.
// Block 256 thr (8 waves) computes 32 full rows (all 256 cols).  Grid 1024.
// Dynamic LDS: Ah 0..2560, Al 2560..5120, Wh 5120..25600, Wl 25600..46080,
//              xrow(f32[32][256]) 46080..78848
// ---------------------------------------------------------------------------
#define PROJ_SMEM 78848

__global__ __launch_bounds__(256) void msa_proj_ln(
    const float* __restrict__ A, const float* __restrict__ W,
    const float* __restrict__ bias, const float* __restrict__ msa,
    const float* __restrict__ gamma, const float* __restrict__ beta,
    float* __restrict__ out)
{
    extern __shared__ char smem[];
    _Float16* Ah = (_Float16*)smem;
    _Float16* Al = (_Float16*)(smem + 2560);
    _Float16* Wh = (_Float16*)(smem + 5120);
    _Float16* Wl = (_Float16*)(smem + 25600);
    float*  xrow = (float*)(smem + 46080);

    const int tid = threadIdx.x;
    const int Lm  = tid & 15;
    const int Lh  = (tid >> 4) & 1;
    const int w   = tid >> 5;
    const int m0  = blockIdx.x * 32;
    const int rt  = w & 1;
    const int ntb = (w >> 1) * 4;

    v8f acc[4];
#pragma unroll
    for (int ct = 0; ct < 4; ++ct) acc[ct] = (v8f){0,0,0,0,0,0,0,0};

    for (int kc = 0; kc < 256; kc += 32) {
        {
            const int arow = tid & 31;
            const int j0   = (tid >> 5) * 4;
            f4v a = *(const f4v*)(A + (size_t)(m0 + arow) * 256 + kc + j0);
#pragma unroll
            for (int j = 0; j < 4; ++j)
                f2hl(a[j], Ah[arow * 40 + j0 + j], Al[arow * 40 + j0 + j]);
        }
        {
            const int nn = tid;
#pragma unroll
            for (int j = 0; j < 32; ++j) {
                float v = W[(size_t)(kc + j) * 256 + nn];
                f2hl(v, Wh[nn * 40 + j], Wl[nn * 40 + j]);
            }
        }
        __syncthreads();

        const _Float16* arp = Ah + (rt * 16 + Lm) * 40 + 8 * Lh;
        const _Float16* alp = Al + (rt * 16 + Lm) * 40 + 8 * Lh;
        v16h fah = ld_a_frag(arp);
        v16h fal = ld_a_frag(alp);
#pragma unroll
        for (int ct = 0; ct < 4; ++ct) {
            const int nt = ntb + ct;
            const _Float16* bhp = Wh + (nt * 16 + Lm) * 40 + 8 * Lh;
            const _Float16* blp = Wl + (nt * 16 + Lm) * 40 + 8 * Lh;
            acc[ct] = wmma3(fah, fal, ld_contig16(bhp), ld_contig16(blp), acc[ct]);
        }
        __syncthreads();
    }

    // bias + residual into LDS row buffer
#pragma unroll
    for (int ct = 0; ct < 4; ++ct) {
        const int n = (ntb + ct) * 16 + Lm;
        const float b = bias[n];
#pragma unroll
        for (int r = 0; r < 8; ++r) {
            const int ml = rt * 16 + r + 8 * Lh;
            xrow[ml * 256 + n] = acc[ct][r] + b + msa[(size_t)(m0 + ml) * 256 + n];
        }
    }
    __syncthreads();

    // LayerNorm: 8 threads per row
    const int row = tid >> 3;
    const int j   = tid & 7;
    float s1 = 0.f, s2 = 0.f;
#pragma unroll
    for (int c = 0; c < 32; ++c) {
        float v = xrow[row * 256 + j * 32 + c];
        s1 += v;
        s2 += v * v;
    }
    s1 += __shfl_xor(s1, 1); s2 += __shfl_xor(s2, 1);
    s1 += __shfl_xor(s1, 2); s2 += __shfl_xor(s2, 2);
    s1 += __shfl_xor(s1, 4); s2 += __shfl_xor(s2, 4);
    const float mu  = s1 * (1.0f / 256.0f);
    const float var = s2 * (1.0f / 256.0f) - mu * mu;
    const float rs  = rsqrtf(var + LN_EPS_C);
#pragma unroll
    for (int c = 0; c < 32; ++c) {
        const int col = j * 32 + c;
        float v = xrow[row * 256 + col];
        out[(size_t)(m0 + row) * 256 + col] = (v - mu) * rs * gamma[col] + beta[col];
    }
}

// ---------------------------------------------------------------------------
// Launch
// ---------------------------------------------------------------------------
extern "C" void kernel_launch(void* const* d_in, const int* in_sizes, int n_in,
                              void* d_out, int out_size, void* d_ws, size_t ws_size,
                              hipStream_t stream) {
    const float* msa   = (const float*)d_in[0];
    const float* Wqkv  = (const float*)d_in[1];
    const float* bqkv  = (const float*)d_in[2];
    const float* Wout  = (const float*)d_in[3];
    const float* bout  = (const float*)d_in[4];
    const float* gamma = (const float*)d_in[5];
    const float* beta  = (const float*)d_in[6];
    const int*   mask  = (const int*)d_in[7];
    float* out = (float*)d_out;

    float* qkv_ws  = (float*)d_ws;                       // 32768*768 f32 = 96MB
    float* attn_ws = qkv_ws + (size_t)32768 * 768;       // 32768*256 f32 = 32MB

    msa_qkv_gemm<<<dim3(256, 12), 256, 0, stream>>>(msa, Wqkv, bqkv, qkv_ws);
    msa_attn<<<1024, 256, ATTN_SMEM, stream>>>(qkv_ws, mask, attn_ws);
    msa_proj_ln<<<1024, 256, PROJ_SMEM, stream>>>(attn_ws, Wout, bout, msa,
                                                  gamma, beta, out);
}
